// TimeAdapter_69827578298972
// MI455X (gfx1250) — matrix operand order repaired and checked by reading, not compiled, not run on hardware
//
#include <hip/hip_runtime.h>

typedef __attribute__((ext_vector_type(16))) __bf16 v16bf;
typedef __attribute__((ext_vector_type(2)))  __bf16 bf16x2;
typedef __attribute__((ext_vector_type(8)))  float  v8f;

namespace {
constexpr int Wd = 256, Hd = 256, Cd = 64;
constexpr int TPX  = 14;   // output pixels produced per wave tile
constexpr int TCOL = 16;   // staged columns per tile: x0-1 .. x0+14
constexpr int WAVES = 2;   // waves per block (independent tiles)
constexpr int XBLK = 10;   // ceil(W / (WAVES*TPX)) block-tiles per row
constexpr int RST  = 36;   // padded row stride (dwords): 16B aligned, conflict-free
constexpr float EPSN  = 1e-12f;
constexpr float ALPHA = -1.0f;
}

struct alignas(16) WaveShm {
  float rawt[3][Cd][TCOL];       // raw nbr rows y-1..y+1            (12 KB)
  float reft[Cd][TCOL];          // raw ref, cols = pixels x0..x0+15 ( 4 KB)
  unsigned bnormT[3][TCOL][RST]; // normalized nbr, packed bf16 pairs, col-major (6.75 KB)
  unsigned refnT[TCOL][RST];     // normalized ref, packed bf16 pairs, col-major (2.25 KB)
  float dmat[3][16][16];         // WMMA D tiles per dy              ( 3 KB)
  float wts[16][9];              // softmax weights
};

__device__ __forceinline__ int refl(int i, int n) {
  i = (i < 0) ? -i : i;
  return (i >= n) ? (2 * n - 2 - i) : i;
}
__device__ __forceinline__ unsigned f2bf(float f) {   // round-to-nearest-even
  union { float f; unsigned u; } x; x.f = f;
  return (x.u + 0x7FFFu + ((x.u >> 16) & 1u)) >> 16;
}
__device__ __forceinline__ unsigned pack2bf(float lo, float hi) {
#if __has_builtin(__builtin_amdgcn_cvt_pk_bf16_f32)
  bf16x2 p = __builtin_amdgcn_cvt_pk_bf16_f32(lo, hi);  // v_cvt_pk_bf16_f32
  return __builtin_bit_cast(unsigned, p);
#else
  return (f2bf(hi) << 16) | f2bf(lo);
#endif
}
__device__ __forceinline__ float bflo(unsigned u) {
  union { unsigned u; float f; } x; x.u = u << 16; return x.f;
}
__device__ __forceinline__ float bfhi(unsigned u) {
  union { unsigned u; float f; } x; x.u = u & 0xFFFF0000u; return x.f;
}
// Wave-level LDS producer->consumer sync (DS ops are in-order per wave).
__device__ __forceinline__ void wave_sync() {
  __builtin_amdgcn_wave_barrier();
  asm volatile("s_wait_dscnt 0" ::: "memory");
  __builtin_amdgcn_wave_barrier();
}

union V16 { uint4 q[2]; v16bf v; };

__global__ __launch_bounds__(WAVES * 32)
void patch_attn_kernel(const float* __restrict__ nbr,
                       const float* __restrict__ ref,
                       float* __restrict__ out) {
  __shared__ WaveShm shm[WAVES];
  const int tid  = threadIdx.x;
  const int wv   = tid >> 5;
  const int lane = tid & 31;
  const int half = lane >> 4;
  const int l16  = lane & 15;

  const int bid = blockIdx.x;
  const int b   = bid / (Hd * XBLK);
  const int rm  = bid % (Hd * XBLK);
  const int y   = rm / XBLK;
  const int xb  = rm % XBLK;
  const int x0  = (xb * WAVES + wv) * TPX;
  if (x0 >= Wd) return;                  // wave-uniform: EXEC stays full
  const int npx = min(TPX, Wd - x0);

  WaveShm& S = shm[wv];

  // ---- stage raw tiles into LDS; accumulate sum-of-squares on the fly ----
  float ssq[3] = {0.f, 0.f, 0.f};  // per (row, col=l16), channels of parity `half`
  float rsq = 0.f;                 // ref, per col=l16, channels of parity `half`
  {
    const float* nb = nbr + (size_t)b * Cd * Hd * Wd;
    const int xx = refl(x0 - 1 + l16, Wd);
    const int yy0 = refl(y - 1, Hd), yy2 = refl(y + 1, Hd);
    const int yrs[3] = {yy0, y, yy2};
    for (int it = 0; it < 96; ++it) {
      int pi = it * 2 + half;            // 0..191 (row,channel) pairs
      int r = pi >> 6, c = pi & 63;
      float v = nb[((size_t)c * Hd + yrs[r]) * Wd + xx];
      S.rawt[r][c][l16] = v;
      ssq[r] += v * v;
    }
    const float* rf = ref + (size_t)b * Cd * Hd * Wd;
    const int xr = refl(x0 + l16, Wd);
    for (int it = 0; it < 32; ++it) {
      int c = it * 2 + half;
      float v = rf[((size_t)c * Hd + y) * Wd + xr];
      S.reft[c][l16] = v;
      rsq += v * v;
    }
  }
  // merge the two channel-parity halves (lane <-> lane^16 share a column)
  float inb[3];
  #pragma unroll
  for (int r = 0; r < 3; ++r) {
    float s = ssq[r] + __shfl_xor(ssq[r], 16, 32);
    inb[r] = 1.0f / fmaxf(sqrtf(s), EPSN);
  }
  const float innr = 1.0f / fmaxf(sqrtf(rsq + __shfl_xor(rsq, 16, 32)), EPSN);
  wave_sync();

  // ---- build packed bf16 operand tiles (col-major, K-pairs contiguous) ----
  for (int it = 0; it < 48; ++it) {
    int p = it * 2 + half;               // 0..95 -> (row, channel-pair)
    int r = p >> 5, cp = p & 31;
    float v0 = S.rawt[r][2 * cp    ][l16] * inb[r];
    float v1 = S.rawt[r][2 * cp + 1][l16] * inb[r];
    S.bnormT[r][l16][cp] = pack2bf(v0, v1);
  }
  for (int it = 0; it < 16; ++it) {
    int cp = it * 2 + half;              // 0..31 channel pairs
    float v0 = S.reft[2 * cp    ][l16] * innr;
    float v1 = S.reft[2 * cp + 1][l16] * innr;
    S.refnT[l16][cp] = pack2bf(v0, v1);
  }
  wave_sync();

  // ---- load A (refn) with wide ds_load_b128 in the ISA 16-bit A layout ----
  // lane M=l16; VGPR 0-3: K = 8*half..8*half+7 ; VGPR 4-7: K = 16+8*half..
  V16 a0, a1;
  a0.q[0] = *(const uint4*)&S.refnT[l16][ 0 + 4 * half];
  a0.q[1] = *(const uint4*)&S.refnT[l16][ 8 + 4 * half];
  a1.q[0] = *(const uint4*)&S.refnT[l16][16 + 4 * half];
  a1.q[1] = *(const uint4*)&S.refnT[l16][24 + 4 * half];

  // ---- 3 dy-rows x (2 x v_wmma_f32_16x16x32_bf16): cosine tiles ----
  // B layout: same as A: VGPR 0-3: K = 8*half..8*half+7 ; VGPR 4-7: K = 16+8*half.. (per chunk)
  #pragma unroll
  for (int dy = 0; dy < 3; ++dy) {
    V16 b0, b1;
    b0.q[0] = *(const uint4*)&S.bnormT[dy][l16][ 0 + 4 * half];
    b0.q[1] = *(const uint4*)&S.bnormT[dy][l16][ 8 + 4 * half];
    b1.q[0] = *(const uint4*)&S.bnormT[dy][l16][16 + 4 * half];
    b1.q[1] = *(const uint4*)&S.bnormT[dy][l16][24 + 4 * half];
    v8f d = {};
    d = __builtin_amdgcn_wmma_f32_16x16x32_bf16(false, a0.v, false, b0.v,
                                                (short)0, d, false, false);
    d = __builtin_amdgcn_wmma_f32_16x16x32_bf16(false, a1.v, false, b1.v,
                                                (short)0, d, false, false);
    #pragma unroll
    for (int r = 0; r < 8; ++r)
      S.dmat[dy][r + 8 * half][l16] = d[r];   // D: lane n=l16, row m=r+8*half
  }
  wave_sync();

  // ---- softmax over the 9 band-diagonal entries D[p, p+dx] ----
  if (lane < npx) {
    const int p = lane;
    float dv[9], mx = -3.0e38f;
    #pragma unroll
    for (int dy = 0; dy < 3; ++dy)
      #pragma unroll
      for (int dx = 0; dx < 3; ++dx) {
        float v = S.dmat[dy][p][p + dx];      // n = p+1+(dx-1)
        dv[dy * 3 + dx] = v;
        mx = fmaxf(mx, v);
      }
    float sum = 0.f;
    #pragma unroll
    for (int k = 0; k < 9; ++k) { dv[k] = __expf(dv[k] - mx); sum += dv[k]; }
    const float inv = 1.0f / sum;
    #pragma unroll
    for (int k = 0; k < 9; ++k) S.wts[p][k] = dv[k] * inv;
  }
  wave_sync();

  // ---- aggregate normalized neighbors (2 channels/iter), wdiff, store ----
  {
    const int p = l16;
    float wt[9];
    if (p < npx) {
      #pragma unroll
      for (int k = 0; k < 9; ++k) wt[k] = S.wts[p][k];
    }
    float* op = out + (((size_t)b * Cd) * Hd + y) * Wd + x0;
    for (int it = 0; it < 16; ++it) {
      int cp = it * 2 + half;              // channel pair
      int c0 = 2 * cp, c1 = c0 + 1;
      if (p < npx) {
        float agg0 = 0.f, agg1 = 0.f;
        #pragma unroll
        for (int dy = 0; dy < 3; ++dy)
          #pragma unroll
          for (int dx = 0; dx < 3; ++dx) {
            unsigned u = S.bnormT[dy][p + dx][cp];
            float w = wt[dy * 3 + dx];
            agg0 += w * bflo(u);
            agg1 += w * bfhi(u);
          }
        float d0 = S.rawt[1][c0][p + 1] - S.reft[c0][p];
        float d1 = S.rawt[1][c1][p + 1] - S.reft[c1][p];
        op[(size_t)c0 * Hd * Wd + p] = agg0 * __expf(ALPHA * d0 * d0);
        op[(size_t)c1 * Hd * Wd + p] = agg1 * __expf(ALPHA * d1 * d1);
      }
    }
  }
}

extern "C" void kernel_launch(void* const* d_in, const int* in_sizes, int n_in,
                              void* d_out, int out_size, void* d_ws, size_t ws_size,
                              hipStream_t stream) {
  (void)n_in; (void)out_size; (void)d_ws; (void)ws_size;
  const float* nbr = (const float*)d_in[0];
  const float* ref = (const float*)d_in[1];
  float* out = (float*)d_out;
  const int nb = in_sizes[0] / (Cd * Hd * Wd);   // batch = 2 for the reference
  dim3 grid(nb * Hd * XBLK);
  dim3 block(WAVES * 32);
  hipLaunchKernelGGL(patch_attn_kernel, grid, block, 0, stream, nbr, ref, out);
}
